// BinocularFormer_44882408243680
// MI455X (gfx1250) — matrix operand order repaired and checked by reading, not compiled, not run on hardware
//
#include <hip/hip_runtime.h>
#include <hip/hip_bf16.h>

typedef __attribute__((ext_vector_type(16))) _Float16 v16h;
typedef __attribute__((ext_vector_type(8)))  _Float16 v8h;
typedef __attribute__((ext_vector_type(8)))  float    v8f;

#if defined(__has_builtin)
#if __has_builtin(__builtin_amdgcn_global_load_async_to_lds_b128)
#define HAVE_ASYNC_LDS 1
#endif
#endif
#ifndef HAVE_ASYNC_LDS
#define HAVE_ASYNC_LDS 0
#endif

#if HAVE_ASYNC_LDS
// builtin signature (from hipcc diagnostic): (int4 AS1*, int4 AS3*, imm, cpol)
typedef int v4i_b128 __attribute__((__vector_size__(16)));
typedef __attribute__((address_space(1))) v4i_b128* gld_ptr;
typedef __attribute__((address_space(3))) v4i_b128* lds_ptr;
#define ASYNC_B128(src, dst) \
    __builtin_amdgcn_global_load_async_to_lds_b128( \
        (gld_ptr)(const void*)(src), (lds_ptr)(void*)(dst), 0, 0)
#if __has_builtin(__builtin_amdgcn_s_wait_asynccnt)
#define WAIT_ASYNC() __builtin_amdgcn_s_wait_asynccnt(0)
#else
#define WAIT_ASYNC() asm volatile("s_wait_asynccnt 0" ::: "memory")
#endif
#endif

__device__ __forceinline__ float gelu_f(float x) {
    const float c0 = 0.7978845608028654f; // sqrt(2/pi)
    float x3 = x * x * x;
    return 0.5f * x * (1.0f + tanhf(c0 * (x + 0.044715f * x3)));
}

// ---------------------------------------------------------------------------
// LDS-staged, double-buffered WMMA GEMM.
//   A: f16 [Mpad, lda] row-major (lda = Kpad, mult of 8; K zero-padded)
//   B: f16 [Npad, ldb] transposed weights B[n][k]
//   C: optional f32 [M,N]; Ch: optional f16 [M,N]
// Block: 256 thr = 8 waves; tile 128M x 64N x 32K. Wave = 32x32 (2x2 WMMA).
// LDS rows stride 40 f16 (80B): ds_load_b128 16B-aligned, conflict-free.
// Fill via async-to-LDS (ASYNCcnt) when available, else load+ds_store.
// A buffers must be allocated with M rounded up to 128 rows (stores guarded).
// ---------------------------------------------------------------------------
#define LDSA (128 * 40)
#define LDSB (64 * 40)

__global__ __launch_bounds__(256) void gemm_f16_wmma(
    const _Float16* __restrict__ A, int lda,
    const _Float16* __restrict__ B, int ldb,
    const float* __restrict__ bias,
    float* __restrict__ C, _Float16* __restrict__ Ch,
    int M, int N, int K, int act)
{
    __shared__ _Float16 As[2][LDSA];
    __shared__ _Float16 Bs[2][LDSB];

    int tid  = threadIdx.x;
    int lane = tid & 31;
    int wave = tid >> 5;
    int wm   = wave >> 1;        // 0..3 -> 32-row slab
    int wn   = wave & 1;         // 0..1 -> 32-col slab
    int hsel = lane >> 4;
    int l16  = lane & 15;
    int bM = blockIdx.y * 128;
    int bN = blockIdx.x * 64;

    // fill indexing: A tile 128x32 (2 chunks/thread), B tile 64x32 (1 chunk)
    int ar0 = (tid) >> 2;            // rows 0..63   (chunk set 0)
    int ar1 = (256 + tid) >> 2;      // rows 64..127 (chunk set 1)
    int ac  = (tid & 3) * 8;
    const _Float16* Ag0 = A + (size_t)(bM + ar0) * lda + ac;
    const _Float16* Ag1 = A + (size_t)(bM + ar1) * lda + ac;
    const _Float16* Bg  = B + (size_t)(bN + ar0) * ldb + ac;

    v8f acc00 = {}, acc01 = {}, acc10 = {}, acc11 = {};

    int nk = K >> 5;

    // ---- prologue: fill buffer 0 ----
#if HAVE_ASYNC_LDS
    ASYNC_B128(Ag0, &As[0][ar0 * 40 + ac]);
    ASYNC_B128(Ag1, &As[0][ar1 * 40 + ac]);
    ASYNC_B128(Bg,  &Bs[0][ar0 * 40 + ac]);
    WAIT_ASYNC();
#else
    {
        v8h a0 = *(const v8h*)(Ag0);
        v8h a1 = *(const v8h*)(Ag1);
        v8h b0 = *(const v8h*)(Bg);
        *(v8h*)(&As[0][ar0 * 40 + ac]) = a0;
        *(v8h*)(&As[0][ar1 * 40 + ac]) = a1;
        *(v8h*)(&Bs[0][ar0 * 40 + ac]) = b0;
    }
#endif
    __syncthreads();

    for (int ki = 0; ki < nk; ++ki) {
        int cur = ki & 1;
        int nxt = cur ^ 1;
        bool hasNext = (ki + 1 < nk);
        int kn = (ki + 1) << 5;

#if HAVE_ASYNC_LDS
        if (hasNext) {
            ASYNC_B128(Ag0 + kn, &As[nxt][ar0 * 40 + ac]);
            ASYNC_B128(Ag1 + kn, &As[nxt][ar1 * 40 + ac]);
            ASYNC_B128(Bg + kn,  &Bs[nxt][ar0 * 40 + ac]);
        }
#else
        v8h nA0, nA1, nB0;
        if (hasNext) {
            nA0 = *(const v8h*)(Ag0 + kn);
            nA1 = *(const v8h*)(Ag1 + kn);
            nB0 = *(const v8h*)(Bg + kn);
        }
#endif

        // ---- compute from current buffer ----
        const _Float16* Asb = &As[cur][0];
        const _Float16* Bsb = &Bs[cur][0];
        v16h af[2], bf[2];
        #pragma unroll
        for (int i = 0; i < 2; ++i) {
            int arow = wm * 32 + i * 16 + l16;
            v8h x0 = *(const v8h*)(Asb + arow * 40 + hsel * 8);
            v8h x1 = *(const v8h*)(Asb + arow * 40 + 16 + hsel * 8);
            af[i] = __builtin_shufflevector(x0, x1, 0,1,2,3,4,5,6,7,8,9,10,11,12,13,14,15);
            int brow = wn * 32 + i * 16 + l16;
            v8h y0 = *(const v8h*)(Bsb + brow * 40 + hsel * 8);
            v8h y1 = *(const v8h*)(Bsb + brow * 40 + 16 + hsel * 8);
            bf[i] = __builtin_shufflevector(y0, y1, 0,1,2,3,4,5,6,7,8,9,10,11,12,13,14,15);
        }
        acc00 = __builtin_amdgcn_wmma_f32_16x16x32_f16(false, af[0], false, bf[0], (short)0, acc00, false, false);
        acc01 = __builtin_amdgcn_wmma_f32_16x16x32_f16(false, af[0], false, bf[1], (short)0, acc01, false, false);
        acc10 = __builtin_amdgcn_wmma_f32_16x16x32_f16(false, af[1], false, bf[0], (short)0, acc10, false, false);
        acc11 = __builtin_amdgcn_wmma_f32_16x16x32_f16(false, af[1], false, bf[1], (short)0, acc11, false, false);

#if HAVE_ASYNC_LDS
        if (hasNext) WAIT_ASYNC();
#else
        if (hasNext) {
            *(v8h*)(&As[nxt][ar0 * 40 + ac]) = nA0;
            *(v8h*)(&As[nxt][ar1 * 40 + ac]) = nA1;
            *(v8h*)(&Bs[nxt][ar0 * 40 + ac]) = nB0;
        }
#endif
        __syncthreads();
    }

    // ---- epilogue: C/D layout -> VGPR r holds row (r + hsel*8), col = l16 ----
    #pragma unroll
    for (int im = 0; im < 2; ++im) {
        #pragma unroll
        for (int in2 = 0; in2 < 2; ++in2) {
            v8f acc = im == 0 ? (in2 == 0 ? acc00 : acc01)
                              : (in2 == 0 ? acc10 : acc11);
            int gcol = bN + wn * 32 + in2 * 16 + l16;
            if (gcol >= N) continue;
            float bv = bias ? bias[gcol] : 0.0f;
            #pragma unroll
            for (int r = 0; r < 8; ++r) {
                int grow = bM + wm * 32 + im * 16 + hsel * 8 + r;
                if (grow >= M) continue;
                float v = acc[r] + bv;
                if (act == 1) v = gelu_f(v);
                if (C)  C[(size_t)grow * N + gcol] = v;
                if (Ch) Ch[(size_t)grow * N + gcol] = (_Float16)v;
            }
        }
    }
}

// weight prep: out[n*Kpad + k] = (transB ? W[n*K+k] : W[k*N+n]), zero-padded
__global__ void prep_weight_kernel(const float* __restrict__ W, _Float16* __restrict__ out,
                                   int K, int N, int Kpad, int Npad, int transB)
{
    int gid = blockIdx.x * 256 + threadIdx.x;
    int total = Npad * Kpad;
    if (gid >= total) return;
    int n = gid / Kpad, k = gid % Kpad;
    float v = 0.0f;
    if (n < N && k < K)
        v = transB ? W[(size_t)n * K + k] : W[(size_t)k * N + n];
    out[gid] = (_Float16)v;
}

// ---------------------------------------------------------------------------
// Conv1: 3->256, 3x3, pad 1, GELU, direct (K=27). out [B*HW, 256] f32.
// ---------------------------------------------------------------------------
__global__ void conv1_kernel(const float* __restrict__ img, const float* __restrict__ W,
                             const float* __restrict__ bias, float* __restrict__ out)
{
    int gid = blockIdx.x * 256 + threadIdx.x;
    if (gid >= 4 * 4096 * 256) return;
    int co = gid & 255;
    int p  = gid >> 8;
    int b  = p >> 12;
    int yx = p & 4095;
    int y  = yx >> 6, x = yx & 63;
    float acc = bias[co];
    #pragma unroll
    for (int i = 0; i < 3; ++i)
        #pragma unroll
        for (int ky = 0; ky < 3; ++ky) {
            int yy = y + ky - 1;
            if (yy < 0 || yy > 63) continue;
            #pragma unroll
            for (int kx = 0; kx < 3; ++kx) {
                int xx = x + kx - 1;
                if (xx < 0 || xx > 63) continue;
                acc += img[((b * 3 + i) * 64 + yy) * 64 + xx] *
                       W[((co * 3 + i) * 3 + ky) * 3 + kx];
            }
        }
    out[(size_t)p * 256 + co] = gelu_f(acc);
}

// im2col -> f16: src [B*HW, 256] f32 -> dst [B*HW, 2304] f16
__global__ void im2col_kernel(const float* __restrict__ src, _Float16* __restrict__ dst)
{
    long gid = (long)blockIdx.x * 256 + threadIdx.x;
    long total = 16384L * 2304L;
    if (gid >= total) return;
    int kcol = (int)(gid % 2304);
    int p    = (int)(gid / 2304);
    int i  = kcol / 9;
    int kk = kcol % 9;
    int ky = kk / 3, kx = kk % 3;
    int b = p >> 12, yx = p & 4095;
    int y = yx >> 6, x = yx & 63;
    int yy = y + ky - 1, xx = x + kx - 1;
    float v = 0.0f;
    if (yy >= 0 && yy < 64 && xx >= 0 && xx < 64)
        v = src[((size_t)(b * 4096 + yy * 64 + xx)) * 256 + i];
    dst[gid] = (_Float16)v;
}

// ---------------------------------------------------------------------------
// Sequential FPS disjoint clustering; 1 block per batch (latency-bound).
// ---------------------------------------------------------------------------
__global__ __launch_bounds__(256) void fps_kernel(
    const float* __restrict__ pos, int n, int cs, int nc,
    int* __restrict__ idx_out, float* __restrict__ mindist,
    float* __restrict__ dist, int* __restrict__ assigned)
{
    int b = blockIdx.x;
    const float* P = pos + (size_t)b * n * 3;
    float* md = mindist + (size_t)b * n;
    float* dd = dist + (size_t)b * n;
    int*   as = assigned + (size_t)b * n;
    int tid = threadIdx.x;
    __shared__ float rv[256];
    __shared__ int   ri[256];
    __shared__ float cpt[3];

    for (int i = tid; i < n; i += 256) { md[i] = 1e30f; as[i] = 0; }
    __syncthreads();

    for (int c = 0; c < nc; ++c) {
        float best = -1e30f; int bi = n;
        for (int i = tid; i < n; i += 256)
            if (!as[i]) {
                float v = md[i];
                if (v > best || (v == best && i < bi)) { best = v; bi = i; }
            }
        rv[tid] = best; ri[tid] = bi;
        __syncthreads();
        for (int off = 128; off > 0; off >>= 1) {
            if (tid < off) {
                if (rv[tid + off] > rv[tid] ||
                    (rv[tid + off] == rv[tid] && ri[tid + off] < ri[tid])) {
                    rv[tid] = rv[tid + off]; ri[tid] = ri[tid + off];
                }
            }
            __syncthreads();
        }
        int center = ri[0];
        if (tid == 0) {
            cpt[0] = P[center * 3]; cpt[1] = P[center * 3 + 1]; cpt[2] = P[center * 3 + 2];
        }
        __syncthreads();
        for (int i = tid; i < n; i += 256) {
            float dx = P[i * 3] - cpt[0];
            float dy = P[i * 3 + 1] - cpt[1];
            float dz = P[i * 3 + 2] - cpt[2];
            dd[i] = dx * dx + dy * dy + dz * dz;
        }
        __syncthreads();
        for (int j = 0; j < cs; ++j) {
            float bmin = 1e30f; int mi = n;
            for (int i = tid; i < n; i += 256)
                if (!as[i]) {
                    float v = dd[i];
                    if (v < bmin || (v == bmin && i < mi)) { bmin = v; mi = i; }
                }
            rv[tid] = bmin; ri[tid] = mi;
            __syncthreads();
            for (int off = 128; off > 0; off >>= 1) {
                if (tid < off) {
                    if (rv[tid + off] < rv[tid] ||
                        (rv[tid + off] == rv[tid] && ri[tid + off] < ri[tid])) {
                        rv[tid] = rv[tid + off]; ri[tid] = ri[tid + off];
                    }
                }
                __syncthreads();
            }
            int sel = ri[0];
            if (tid == 0) {
                idx_out[((size_t)b * nc + c) * cs + j] = sel;
                as[sel] = 1;
            }
            __syncthreads();
        }
        for (int i = tid; i < n; i += 256) md[i] = fminf(md[i], dd[i]);
        __syncthreads();
    }
}

__global__ void gather_pos_kernel(const float* __restrict__ pos, const int* __restrict__ idx,
                                  float* __restrict__ pos_g, int nc, int cs, int nPts, int total)
{
    int gid = blockIdx.x * 256 + threadIdx.x;
    if (gid >= total) return;
    int a = gid % 3;
    int e = gid / 3;
    int m = e / cs, t = e % cs;
    int b = m / nc;
    int src = idx[(size_t)m * cs + t];
    pos_g[gid] = pos[((size_t)b * nPts + src) * 3 + a];
}

// tokens: x (f32 residual base) and xh (f16 GEMM A). CLS at t==cs.
__global__ void build_tokens_kernel(const float* __restrict__ feat, const float* __restrict__ pos,
                                    const int* __restrict__ idx, const float* __restrict__ posW,
                                    const float* __restrict__ posBv, const float* __restrict__ cls,
                                    float* __restrict__ x, _Float16* __restrict__ xh,
                                    int nc, int cs, int nPts, int total)
{
    int gid = blockIdx.x * 256 + threadIdx.x;
    if (gid >= total) return;
    int ch  = gid & 255;
    int row = gid >> 8;
    int T = cs + 1;
    int m = row / T, t = row - m * T;
    float v;
    if (t == cs) {
        v = cls[ch];
    } else {
        int b = m / nc;
        int src = idx[(size_t)m * cs + t];
        const float* pp = pos + ((size_t)b * nPts + src) * 3;
        v = feat[((size_t)b * nPts + src) * 256 + ch]
            + pp[0] * posW[ch] + pp[1] * posW[256 + ch] + pp[2] * posW[512 + ch]
            + posBv[ch];
    }
    x[(size_t)row * 256 + ch] = v;
    xh[(size_t)row * 256 + ch] = (_Float16)v;
}

// per-(cluster,head) attention over T=5 tokens, hd=64; f16 output
__global__ __launch_bounds__(64) void attn_small(const float* __restrict__ qkv,
                                                 _Float16* __restrict__ out, int T)
{
    int m = blockIdx.x >> 2;
    int h = blockIdx.x & 3;
    int d = threadIdx.x;  // 0..63
    __shared__ float q[5][64], k[5][64], v[5][64], sc[5][5], att[5][5];
    for (int t = 0; t < T; ++t) {
        size_t rb = ((size_t)(m * T + t)) * 768 + h * 64 + d;
        q[t][d] = qkv[rb];
        k[t][d] = qkv[rb + 256];
        v[t][d] = qkv[rb + 512];
    }
    __syncthreads();
    if (d < T * T) {
        int t = d / T, s = d % T;
        float acc = 0.0f;
        #pragma unroll
        for (int i = 0; i < 64; ++i) acc += q[t][i] * k[s][i];
        sc[t][s] = acc * 0.125f;
    }
    __syncthreads();
    if (d < T) {
        float mx = -1e30f;
        for (int s = 0; s < T; ++s) mx = fmaxf(mx, sc[d][s]);
        float sum = 0.0f;
        for (int s = 0; s < T; ++s) { float e = expf(sc[d][s] - mx); att[d][s] = e; sum += e; }
        float inv = 1.0f / sum;
        for (int s = 0; s < T; ++s) att[d][s] *= inv;
    }
    __syncthreads();
    for (int t = 0; t < T; ++t) {
        float acc = 0.0f;
        for (int s = 0; s < T; ++s) acc += att[t][s] * v[s][d];
        out[((size_t)(m * T + t)) * 256 + h * 64 + d] = (_Float16)acc;
    }
}

// x = LayerNorm(x + y); also writes f16 copy
__global__ __launch_bounds__(64) void add_ln(float* __restrict__ x, const float* __restrict__ y,
                                             const float* __restrict__ g, const float* __restrict__ b,
                                             _Float16* __restrict__ xh)
{
    int row = blockIdx.x;
    int tid = threadIdx.x;
    __shared__ float red[64];
    float vals[4];
    float s = 0.0f;
    #pragma unroll
    for (int i = 0; i < 4; ++i) {
        int c = tid + i * 64;
        float v = x[(size_t)row * 256 + c] + y[(size_t)row * 256 + c];
        vals[i] = v; s += v;
    }
    red[tid] = s; __syncthreads();
    for (int off = 32; off > 0; off >>= 1) {
        if (tid < off) red[tid] += red[tid + off];
        __syncthreads();
    }
    float mean = red[0] * (1.0f / 256.0f);
    __syncthreads();
    float s2 = 0.0f;
    #pragma unroll
    for (int i = 0; i < 4; ++i) { float dlt = vals[i] - mean; s2 += dlt * dlt; }
    red[tid] = s2; __syncthreads();
    for (int off = 32; off > 0; off >>= 1) {
        if (tid < off) red[tid] += red[tid + off];
        __syncthreads();
    }
    float rstd = rsqrtf(red[0] * (1.0f / 256.0f) + 1e-5f);
    #pragma unroll
    for (int i = 0; i < 4; ++i) {
        int c = tid + i * 64;
        float v = (vals[i] - mean) * rstd * g[c] + b[c];
        x[(size_t)row * 256 + c] = v;
        xh[(size_t)row * 256 + c] = (_Float16)v;
    }
}

__global__ void summary_kernel(const float* __restrict__ x, float* __restrict__ featNext,
                               int T, int total)
{
    int gid = blockIdx.x * 256 + threadIdx.x;
    if (gid >= total) return;
    int ch = gid & 255;
    int m  = gid >> 8;
    featNext[gid] = x[((size_t)m * T + (T - 1)) * 256 + ch];
}

__global__ void centroid_kernel(const float* __restrict__ pos_g, float* __restrict__ posNext,
                                int cs, int total)
{
    int gid = blockIdx.x * 256 + threadIdx.x;
    if (gid >= total) return;
    int a = gid % 3;
    int m = gid / 3;
    float s = 0.0f;
    for (int t = 0; t < cs; ++t) s += pos_g[((size_t)m * cs + t) * 3 + a];
    posNext[gid] = s / (float)cs;
}

// head input f16 [128, 288] (rows 64..127 padding): concat(pos, feat), K-pad
__global__ void build_headin_kernel(const float* __restrict__ pos, const float* __restrict__ feat,
                                    _Float16* __restrict__ inp, int total)
{
    int gid = blockIdx.x * 256 + threadIdx.x;
    if (gid >= total) return;
    int c = gid % 288;
    int r = gid / 288;
    float v = 0.0f;
    if (r < 64) {
        if (c < 3) v = pos[r * 3 + c];
        else if (c < 259) v = feat[(size_t)r * 256 + (c - 3)];
    }
    inp[gid] = (_Float16)v;
}

// d_out[0:5120] = [nc=16][heads=4][B=4][C=20]; d_out[5120:5312] = pos [B,16,3]
__global__ void stack_out_kernel(const float* __restrict__ headout, const float* __restrict__ pos,
                                 float* __restrict__ out, int total)
{
    int gid = blockIdx.x * 256 + threadIdx.x;
    if (gid >= total) return;
    if (gid < 5120) {
        int cl = gid % 20;
        int rest = gid / 20;
        int b = rest % 4;
        int rest2 = rest / 4;
        int j = rest2 % 4;
        int c = rest2 / 4;
        out[gid] = headout[(size_t)j * 64 * 20 + (b * 16 + c) * 20 + cl];
    } else {
        out[gid] = pos[gid - 5120];
    }
}

// ---------------------------------------------------------------------------
// Host orchestration
// ---------------------------------------------------------------------------
extern "C" void kernel_launch(void* const* d_in, const int* in_sizes, int n_in,
                              void* d_out, int out_size, void* d_ws, size_t ws_size,
                              hipStream_t stream)
{
    (void)in_sizes; (void)n_in; (void)out_size; (void)ws_size;

    const float* positions = (const float*)d_in[0];
    const float* image     = (const float*)d_in[1];
    const float* cnnW0 = (const float*)d_in[2];
    const float* cnnB0 = (const float*)d_in[3];
    const float* cnnW1 = (const float*)d_in[4];
    const float* cnnB1 = (const float*)d_in[5];
    auto P = [&](int i) { return (const float*)d_in[i]; };

    char* wsp = (char*)d_ws;
    auto carve = [&](size_t bytes) {
        char* p = wsp;
        wsp += (bytes + 255) & ~(size_t)255;
        return (void*)p;
    };
    float*    feat0   = (float*)carve((size_t)16384 * 256 * 4);
    float*    featB   = (float*)carve((size_t)4096 * 256 * 4);
    float*    convA   = (float*)carve((size_t)16384 * 256 * 4);
    _Float16* colh    = (_Float16*)carve((size_t)16384 * 2304 * 2);
    _Float16* cnnW1h  = (_Float16*)carve((size_t)256 * 2304 * 2);
    float*    posA    = (float*)carve((size_t)4 * 4096 * 3 * 4);
    float*    posB2   = (float*)carve((size_t)4 * 4096 * 3 * 4);
    float*    pos_g   = (float*)carve((size_t)16384 * 3 * 4);
    int*      idx     = (int*)carve((size_t)4 * 1024 * 4 * 4);
    float*    mind    = (float*)carve((size_t)4 * 4096 * 4);
    float*    dist    = (float*)carve((size_t)4 * 4096 * 4);
    int*      assigned= (int*)carve((size_t)4 * 4096 * 4);
    float*    xtok    = (float*)carve((size_t)20480 * 256 * 4);
    _Float16* xtok_h  = (_Float16*)carve((size_t)20480 * 256 * 2);
    float*    qkv     = (float*)carve((size_t)20480 * 768 * 4);
    _Float16* attno_h = (_Float16*)carve((size_t)20480 * 256 * 2);
    _Float16* ffnh_h  = (_Float16*)carve((size_t)20480 * 512 * 2);
    float*    tmp2    = (float*)carve((size_t)20480 * 256 * 4);
    _Float16* Wqkvh   = (_Float16*)carve((size_t)768 * 256 * 2);
    _Float16* Woh     = (_Float16*)carve((size_t)256 * 256 * 2);
    _Float16* W1h     = (_Float16*)carve((size_t)512 * 256 * 2);
    _Float16* W2h     = (_Float16*)carve((size_t)256 * 512 * 2);
    _Float16* headin_h= (_Float16*)carve((size_t)128 * 288 * 2);
    _Float16* headh_h = (_Float16*)carve((size_t)128 * 256 * 2);
    _Float16* hW1h    = (_Float16*)carve((size_t)256 * 288 * 2);
    _Float16* hW2h    = (_Float16*)carve((size_t)64 * 256 * 2);
    float*    headout = (float*)carve((size_t)4 * 64 * 20 * 4);

    // ---- CNN ----
    conv1_kernel<<<(4 * 4096 * 256 + 255) / 256, 256, 0, stream>>>(image, cnnW0, cnnB0, convA);
    {
        long total = 16384L * 2304L;
        im2col_kernel<<<(int)((total + 255) / 256), 256, 0, stream>>>(convA, colh);
        prep_weight_kernel<<<(256 * 2304 + 255) / 256, 256, 0, stream>>>(
            cnnW1, cnnW1h, 2304, 256, 2304, 256, 1);
        dim3 g(256 / 64, 16384 / 128);
        gemm_f16_wmma<<<g, 256, 0, stream>>>(colh, 2304, cnnW1h, 2304, cnnB1,
                                             feat0, (_Float16*)nullptr,
                                             16384, 256, 2304, 1);
    }

    // ---- hierarchical stages ----
    const float* posCur = positions;
    float* featCur = feat0;
    int nPts = 4096;
    float* posBufs[2]  = { posA, posB2 };
    float* featBufs[2] = { featB, feat0 };

    for (int s = 0; s < 4; ++s) {
        const int cs = 4;
        int nc = nPts / cs;
        int Mc = 4 * nc;
        int T  = cs + 1;
        int R  = Mc * T;
        int Ry = (R + 127) / 128;
        int sb = 6 + s * 27;
        const float* posW  = P(sb + 0);
        const float* posBb = P(sb + 1);
        const float* cls   = P(sb + 2);

        fps_kernel<<<4, 256, 0, stream>>>(posCur, nPts, cs, nc, idx, mind, dist, assigned);
        gather_pos_kernel<<<(Mc * cs * 3 + 255) / 256, 256, 0, stream>>>(
            posCur, idx, pos_g, nc, cs, nPts, Mc * cs * 3);
        build_tokens_kernel<<<(R * 256 + 255) / 256, 256, 0, stream>>>(
            featCur, posCur, idx, posW, posBb, cls, xtok, xtok_h, nc, cs, nPts, R * 256);

        for (int l = 0; l < 2; ++l) {
            int lb = sb + 3 + l * 12;
            const float* Wqkv = P(lb + 0); const float* bqkv = P(lb + 1);
            const float* Wo   = P(lb + 2); const float* bo   = P(lb + 3);
            const float* ln1g = P(lb + 4); const float* ln1b = P(lb + 5);
            const float* W1   = P(lb + 6); const float* b1   = P(lb + 7);
            const float* W2   = P(lb + 8); const float* b2   = P(lb + 9);
            const float* ln2g = P(lb + 10); const float* ln2b = P(lb + 11);

            prep_weight_kernel<<<(768 * 256 + 255) / 256, 256, 0, stream>>>(
                Wqkv, Wqkvh, 256, 768, 256, 768, 0);
            prep_weight_kernel<<<(256 * 256 + 255) / 256, 256, 0, stream>>>(
                Wo, Woh, 256, 256, 256, 256, 0);
            prep_weight_kernel<<<(512 * 256 + 255) / 256, 256, 0, stream>>>(
                W1, W1h, 256, 512, 256, 512, 0);
            prep_weight_kernel<<<(256 * 512 + 255) / 256, 256, 0, stream>>>(
                W2, W2h, 512, 256, 512, 256, 0);

            { dim3 g(768 / 64, Ry);
              gemm_f16_wmma<<<g, 256, 0, stream>>>(xtok_h, 256, Wqkvh, 256, bqkv,
                                                   qkv, (_Float16*)nullptr, R, 768, 256, 0); }
            attn_small<<<Mc * 4, 64, 0, stream>>>(qkv, attno_h, T);
            { dim3 g(256 / 64, Ry);
              gemm_f16_wmma<<<g, 256, 0, stream>>>(attno_h, 256, Woh, 256, bo,
                                                   tmp2, (_Float16*)nullptr, R, 256, 256, 0); }
            add_ln<<<R, 64, 0, stream>>>(xtok, tmp2, ln1g, ln1b, xtok_h);
            { dim3 g(512 / 64, Ry);
              gemm_f16_wmma<<<g, 256, 0, stream>>>(xtok_h, 256, W1h, 256, b1,
                                                   (float*)nullptr, ffnh_h, R, 512, 256, 1); }
            { dim3 g(256 / 64, Ry);
              gemm_f16_wmma<<<g, 256, 0, stream>>>(ffnh_h, 512, W2h, 512, b2,
                                                   tmp2, (_Float16*)nullptr, R, 256, 512, 0); }
            add_ln<<<R, 64, 0, stream>>>(xtok, tmp2, ln2g, ln2b, xtok_h);
        }

        float* featNext = featBufs[s & 1];
        float* posNext  = posBufs[s & 1];
        summary_kernel<<<(Mc * 256 + 255) / 256, 256, 0, stream>>>(xtok, featNext, T, Mc * 256);
        centroid_kernel<<<(Mc * 3 + 255) / 256, 256, 0, stream>>>(pos_g, posNext, cs, Mc * 3);
        featCur = featNext;
        posCur  = posNext;
        nPts = nc;
    }

    // ---- detection heads ----
    build_headin_kernel<<<(128 * 288 + 255) / 256, 256, 0, stream>>>(posCur, featCur, headin_h, 128 * 288);
    for (int j = 0; j < 4; ++j) {
        int hb = 114 + j * 4;  // 6 + 4*27
        prep_weight_kernel<<<(256 * 288 + 255) / 256, 256, 0, stream>>>(
            P(hb + 0), hW1h, 259, 256, 288, 256, 0);
        prep_weight_kernel<<<(64 * 256 + 255) / 256, 256, 0, stream>>>(
            P(hb + 2), hW2h, 256, 20, 256, 64, 0);
        { dim3 g(256 / 64, 1);
          gemm_f16_wmma<<<g, 256, 0, stream>>>(headin_h, 288, hW1h, 288, P(hb + 1),
                                               (float*)nullptr, headh_h, 64, 256, 288, 1); }
        { dim3 g(1, 1);
          gemm_f16_wmma<<<g, 256, 0, stream>>>(headh_h, 256, hW2h, 256, P(hb + 3),
                                               headout + (size_t)j * 64 * 20, (_Float16*)nullptr,
                                               64, 20, 256, 0); }
    }
    stack_out_kernel<<<(5312 + 255) / 256, 256, 0, stream>>>(headout, posCur, (float*)d_out, 5312);
}
